// MirasModel_48515950575726
// MI455X (gfx1250) — matrix operand order repaired and checked by reading, not compiled, not run on hardware
//
#include <hip/hip_runtime.h>
#include <hip/hip_bf16.h>
#include <math.h>

#define TSEQ   64
#define DMODEL 3136
#define HDIM   512
#define EPSN   1e-6f
#define BN     64     // N-cols per block in the TDM-staged GEMM

typedef __attribute__((ext_vector_type(16))) __bf16 bf16x16;
typedef __attribute__((ext_vector_type(8)))  float  floatx8;
typedef __attribute__((ext_vector_type(4)))  unsigned int tdm_u32x4;
typedef __attribute__((ext_vector_type(8)))  int          tdm_i32x8;
typedef __attribute__((ext_vector_type(4)))  int          tdm_i32x4;

#if defined(__AMDGCN__) && __has_builtin(__builtin_amdgcn_tensor_load_to_lds)
#define HAVE_TDM 1
#else
#define HAVE_TDM 0
#endif

// ---------------------------------------------------------------------------
// WMMA fragment loaders — CDNA5 v_wmma_f32_16x16x32_bf16 layouts (ISA 7.12.2)
// A 16(M)x32(K):  lane l -> m=l%16; lane-half adds K+8; VGPR p holds K pair
// B 32(K)x16(N):  lane l -> n=l%16; lane-half adds K+8; VGPR p holds K pair (same as A)
// C/D 16x16 f32:  lane l -> n=l%16; VGPR v -> m = v + (l>=16 ? 8 : 0)
// fp32 data converted to bf16 in-register (weights are consumed once per call,
// so direct fp32 reads are the bandwidth-optimal path on a 23.3 TB/s part).
// ---------------------------------------------------------------------------
template<bool ATRANS>
__device__ inline bf16x16 load_a_frag(const float* __restrict__ A, int lda,
                                      int m, int kb, int lane)
{
  bf16x16 a;
  const int khalf = (lane >= 16) ? 8 : 0;
#pragma unroll
  for (int p = 0; p < 8; ++p) {
    const int kk = kb + khalf + ((p < 4) ? (2 * p) : (16 + 2 * (p - 4)));
    float x0, x1;
    if (ATRANS) {                       // A stored (K x M): A[k*lda + m]
      x0 = A[(size_t)kk * lda + m];
      x1 = A[(size_t)(kk + 1) * lda + m];
    } else {                            // A stored (M x K): contiguous K pair
      const float2 v = *reinterpret_cast<const float2*>(A + (size_t)m * lda + kk);
      x0 = v.x; x1 = v.y;
    }
    a[2 * p]     = (__bf16)x0;
    a[2 * p + 1] = (__bf16)x1;
  }
  return a;
}

template<bool BTRANS>
__device__ inline bf16x16 load_b_frag(const float* __restrict__ B, int ldb,
                                      int n, int kb, int lane)
{
  bf16x16 b;
  const int khalf = (lane >= 16) ? 8 : 0;
#pragma unroll
  for (int p = 0; p < 8; ++p) {
    const int kk = kb + khalf + ((p < 4) ? (2 * p) : (16 + 2 * (p - 4)));
    float x0, x1;
    if (BTRANS) {                       // B stored (N x K): contiguous K pair
      const float2 v = *reinterpret_cast<const float2*>(B + (size_t)n * ldb + kk);
      x0 = v.x; x1 = v.y;
    } else {                            // B stored (K x N): B[k*ldb + n]
      x0 = B[(size_t)kk * ldb + n];
      x1 = B[(size_t)(kk + 1) * ldb + n];
    }
    b[2 * p]     = (__bf16)x0;
    b[2 * p + 1] = (__bf16)x1;
  }
  return b;
}

// ---------------------------------------------------------------------------
// Tensor Data Mover staging (CDNA5 D# descriptor, ISA cdna5 §8.3-8.4)
// ---------------------------------------------------------------------------
__device__ inline void wait_tensorcnt0()
{
#if defined(__AMDGCN__) && __has_builtin(__builtin_amdgcn_s_wait_tensorcnt)
  __builtin_amdgcn_s_wait_tensorcnt(0);
#else
  asm volatile("s_wait_tensorcnt 0x0" ::: "memory");
#endif
}

#if HAVE_TDM
// Issue one 2-D tile load global->LDS: tile_rows x tile_cols f32 elements,
// row stride `stride_elems` (in elements). TDM ignores EXEC; issue from one lane.
__device__ inline void tdm_issue_2d(unsigned lds_off, const float* gtile,
                                    unsigned tile_cols, unsigned tile_rows,
                                    unsigned long long stride_elems)
{
  const unsigned long long ga = (unsigned long long)(uintptr_t)gtile;
  // Group0: [1:0]=count=1 | [63:32]=lds_addr | [120:64]=global_addr | [127:126]=type(2)
  tdm_u32x4 g0 = { 1u, lds_off, (unsigned)ga,
                   (unsigned)((ga >> 32) & 0x01FFFFFFu) | (2u << 30) };
  // Group1: [17:16]=data_size(2 -> 4B); [79:48]=tensor_dim0; [111:80]=tensor_dim1;
  //         [127:112]=tile_dim0; [143:128]=tile_dim1; [207:160]=tensor_dim0_stride
  const unsigned d0 = tile_cols, d1 = tile_rows;
  const unsigned w1 = (d0 & 0xFFFFu) << 16;
  const unsigned w2 = (d0 >> 16) | ((d1 & 0xFFFFu) << 16);
  const unsigned w3 = (d1 >> 16) | ((tile_cols & 0xFFFFu) << 16);
  const unsigned w4 = tile_rows & 0xFFFFu;
  const unsigned w5 = (unsigned)(stride_elems & 0xFFFFFFFFull);
  const unsigned w6 = (unsigned)((stride_elems >> 32) & 0xFFFFull);
  tdm_i32x8 g1 = { (int)(2u << 16), (int)w1, (int)w2, (int)w3,
                   (int)w4, (int)w5, (int)w6, 0 };
  tdm_i32x4 z4 = { 0, 0, 0, 0 };
#if __clang_major__ >= 23
  tdm_i32x8 z8 = { 0, 0, 0, 0, 0, 0, 0, 0 };
  __builtin_amdgcn_tensor_load_to_lds(g0, g1, z4, z4, z8, 0);
#else
  __builtin_amdgcn_tensor_load_to_lds(g0, g1, z4, z4, 0);
#endif
}
#endif

// Stage A(64 x 32 @lda, row kb offset) and B(32 x BN @ldb, origin kb,n0) tiles.
__device__ inline void stage_tiles(float* At, const float* A, int lda,
                                   float* Bt, const float* B, int ldb,
                                   int kb, int n0)
{
#if HAVE_TDM
  if (threadIdx.x == 0) {
    tdm_issue_2d((unsigned)(uintptr_t)At, A + kb, 32u, 64u, (unsigned long long)lda);
    tdm_issue_2d((unsigned)(uintptr_t)Bt, B + (size_t)kb * ldb + n0, (unsigned)BN, 32u,
                 (unsigned long long)ldb);
  }
#else
  const int tid = threadIdx.x;
  for (int i = tid; i < 64 * 32 / 2; i += 128) {
    const int r = (2 * i) / 32, c = (2 * i) % 32;
    *reinterpret_cast<float2*>(&At[r * 32 + c]) =
        *reinterpret_cast<const float2*>(&A[(size_t)r * lda + kb + c]);
  }
  for (int i = tid; i < 32 * BN / 2; i += 128) {
    const int r = (2 * i) / BN, c = (2 * i) % BN;
    *reinterpret_cast<float2*>(&Bt[r * BN + c]) =
        *reinterpret_cast<const float2*>(&B[(size_t)(kb + r) * ldb + n0 + c]);
  }
#endif
}

__device__ inline void stage_wait(int wave)
{
#if HAVE_TDM
  if (wave == 0) wait_tensorcnt0();
#else
  (void)wave;
#endif
}

// ---------------------------------------------------------------------------
// TDM-staged NN GEMM:  C(64 x N) = A(64 x K) @ B(K x N) + bias
// One block = 4 waves = 64x64 C tile; wave w owns rows 16w..16w+15 and
// reuses one A fragment across 4 WMMAs. Double-buffered LDS tiles fed by
// the Tensor Data Mover, overlapped with WMMA compute (TENSORcnt wait after
// compute, before the barrier). blockIdx.z batches two independent GEMMs
// (keys & vals run concurrently to keep more HBM requests in flight).
// Requires: M == 64, K % 32 == 0, N % 64 == 0.
// ---------------------------------------------------------------------------
__global__ void __launch_bounds__(128)
gemm_nn_tdm(const float* __restrict__ A0, const float* __restrict__ B0,
            const float* __restrict__ bias0, float* __restrict__ C0,
            const float* __restrict__ A1, const float* __restrict__ B1,
            const float* __restrict__ bias1, float* __restrict__ C1,
            int N, int K, int lda, int ldb, int ldc)
{
  __shared__ float Atile[2][64 * 32];   // 8 KB x2
  __shared__ float Btile[2][32 * BN];   // 8 KB x2

  const int z = blockIdx.z;
  const float* A    = z ? A1 : A0;
  const float* B    = z ? B1 : B0;
  const float* bias = z ? bias1 : bias0;
  float*       C    = z ? C1 : C0;

  const int n0   = blockIdx.x * BN;
  const int lane = threadIdx.x & 31;
  const int wave = threadIdx.x >> 5;
  const int nsteps = K / 32;

  floatx8 acc[4] = {};

  stage_tiles(Atile[0], A, lda, Btile[0], B, ldb, 0, n0);
  stage_wait(wave);
  __syncthreads();

  for (int s = 0; s < nsteps; ++s) {
    const int cur = s & 1;
    if (s + 1 < nsteps)
      stage_tiles(Atile[cur ^ 1], A, lda, Btile[cur ^ 1], B, ldb, (s + 1) * 32, n0);

    const float* At = Atile[cur];
    const float* Bt = Btile[cur];
    const bf16x16 a = load_a_frag<false>(At, 32, wave * 16 + (lane & 15), 0, lane);
#pragma unroll
    for (int j = 0; j < 4; ++j) {
      const bf16x16 b = load_b_frag<false>(Bt, BN, j * 16 + (lane & 15), 0, lane);
      acc[j] = __builtin_amdgcn_wmma_f32_16x16x32_bf16(
          false, a, false, b, (short)0, acc[j], false, false);
    }

    if (s + 1 < nsteps) stage_wait(wave);
    __syncthreads();
  }

#pragma unroll
  for (int j = 0; j < 4; ++j) {
    const int n = n0 + j * 16 + (lane & 15);
    const float bn = bias[n];
#pragma unroll
    for (int v = 0; v < 8; ++v) {
      const int m = wave * 16 + v + ((lane >= 16) ? 8 : 0);
      C[(size_t)m * ldc + n] = acc[j][v] + bn;
    }
  }
}

// ---------------------------------------------------------------------------
// Register-direct WMMA GEMM for the small/transposed grad GEMMs (K=64 etc.)
// ---------------------------------------------------------------------------
template<bool ATRANS, bool BTRANS>
__global__ void gemm_wmma_bf16(const float* __restrict__ A,
                               const float* __restrict__ B,
                               const float* __restrict__ bias,
                               float* __restrict__ C,
                               int M, int N, int K, int lda, int ldb, int ldc)
{
  const int lane = threadIdx.x & 31;
  const int wave = threadIdx.x >> 5;
  const int tn   = blockIdx.x * 4 + wave;
  const int tm   = blockIdx.y;
  if (tn * 16 >= N) return;                   // wave-uniform: EXEC stays all-1s

  const int m0 = tm * 16;
  const int n0 = tn * 16;
  const int mA = m0 + (lane & 15);
  const int nB = n0 + (lane & 15);

  floatx8 acc = {};
  for (int kb = 0; kb < K; kb += 32) {
    bf16x16 a = load_a_frag<ATRANS>(A, lda, mA, kb, lane);
    bf16x16 b = load_b_frag<BTRANS>(B, ldb, nB, kb, lane);
    acc = __builtin_amdgcn_wmma_f32_16x16x32_bf16(
        false, a, false, b, (short)0, acc, false, false);
  }

  const float bn = bias ? bias[nB] : 0.0f;
#pragma unroll
  for (int v = 0; v < 8; ++v) {
    const int m = m0 + v + ((lane >= 16) ? 8 : 0);
    C[(size_t)m * ldc + nB] = acc[v] + bn;
  }
  (void)M;
}

// ---------------------------------------------------------------------------
// Elementwise / reduction helpers
// ---------------------------------------------------------------------------
__device__ inline float gelu_tanh(float x) {
  const float c = 0.7978845608028654f;           // sqrt(2/pi)
  const float t = c * (x + 0.044715f * x * x * x);
  return 0.5f * x * (1.0f + tanhf(t));
}
__device__ inline float gelu_tanh_grad(float x) {
  const float c  = 0.7978845608028654f;
  const float x2 = x * x;
  const float t  = c * (x + 0.044715f * x * x2);
  const float th = tanhf(t);
  const float dt = c * (1.0f + 3.0f * 0.044715f * x2);
  return 0.5f * (1.0f + th) + 0.5f * x * (1.0f - th * th) * dt;
}

__global__ void conv_rms_kernel(const float* __restrict__ x,
                                const float* __restrict__ wk, const float* __restrict__ bk,
                                const float* __restrict__ sck,
                                const float* __restrict__ wv, const float* __restrict__ bv,
                                const float* __restrict__ scv,
                                float* __restrict__ nk, float* __restrict__ nv)
{
  const int t = blockIdx.x;
  const float* xt = x + (size_t)t * DMODEL;     // (4,28,28)
  for (int pix = threadIdx.x; pix < 784; pix += blockDim.x) {
    const int h = pix / 28, w = pix % 28;
    float ak[4] = {0.f, 0.f, 0.f, 0.f};
    float av[4] = {0.f, 0.f, 0.f, 0.f};
    for (int kh = 0; kh < 3; ++kh) {
      const int hh = h + kh - 1;
      if (hh < 0 || hh >= 28) continue;
      for (int kw = 0; kw < 3; ++kw) {
        const int ww = w + kw - 1;
        if (ww < 0 || ww >= 28) continue;
        for (int ci = 0; ci < 4; ++ci) {
          const float xv   = xt[ci * 784 + hh * 28 + ww];
          const int   base = ((kh * 3 + kw) * 4 + ci) * 4;
#pragma unroll
          for (int co = 0; co < 4; ++co) {
            ak[co] += xv * wk[base + co];
            av[co] += xv * wv[base + co];
          }
        }
      }
    }
    float ssk = 0.f, ssv = 0.f;
#pragma unroll
    for (int co = 0; co < 4; ++co) {
      ak[co] += bk[co]; av[co] += bv[co];
      ssk += ak[co] * ak[co]; ssv += av[co] * av[co];
    }
    const float ik = rsqrtf(ssk * 0.25f + EPSN);
    const float iv = rsqrtf(ssv * 0.25f + EPSN);
#pragma unroll
    for (int co = 0; co < 4; ++co) {
      nk[(size_t)t * DMODEL + pix * 4 + co] = ak[co] * ik * sck[co];
      nv[(size_t)t * DMODEL + pix * 4 + co] = av[co] * iv * scv[co];
    }
  }
}

__global__ void gelu_kernel(const float* __restrict__ in, float* __restrict__ out, int n) {
  const int i = blockIdx.x * blockDim.x + threadIdx.x;
  if (i < n) out[i] = gelu_tanh(in[i]);
}
__global__ void dgelu_mul_kernel(const float* __restrict__ dH,
                                 const float* __restrict__ Hpre,
                                 float* __restrict__ dG, int n) {
  const int i = blockIdx.x * blockDim.x + threadIdx.x;
  if (i < n) dG[i] = dH[i] * gelu_tanh_grad(Hpre[i]);
}
__global__ void sub_kernel(const float* __restrict__ p, const float* __restrict__ g,
                           float* __restrict__ out, int n) {
  const int i = blockIdx.x * blockDim.x + threadIdx.x;
  if (i < n) out[i] = p[i] - g[i];
}
__global__ void colsum_kernel(const float* __restrict__ Mx, float* __restrict__ out,
                              int rows, int N) {
  const int n = blockIdx.x * blockDim.x + threadIdx.x;
  if (n >= N) return;
  float s = 0.f;
  for (int r = 0; r < rows; ++r) s += Mx[(size_t)r * N + n];
  out[n] = s;
}

__device__ inline float block_reduce_256(float v, float* red) {
  red[threadIdx.x] = v;
  __syncthreads();
  for (int st = 128; st > 0; st >>= 1) {
    if ((int)threadIdx.x < st) red[threadIdx.x] += red[threadIdx.x + st];
    __syncthreads();
  }
  const float r = red[0];
  __syncthreads();
  return r;
}

// Backward through rmsnorm + L2 loss -> weighted dY and weighted scale grad.
// wgt includes 1e-4 * ETA0 * alpha^(T-1) (the per-token weight is constant).
__global__ void rms_bwd_kernel(const float* __restrict__ Ypre,
                               const float* __restrict__ vals,
                               const float* __restrict__ sc,
                               float* __restrict__ dYw, float* __restrict__ dSc,
                               float wgt)
{
  __shared__ float red[256];
  const int t = blockIdx.x;
  const float* y = Ypre + (size_t)t * DMODEL;
  const float* v = vals + (size_t)t * DMODEL;

  float s = 0.f;
  for (int j = threadIdx.x; j < DMODEL; j += 256) { const float a = y[j]; s += a * a; }
  const float inv = rsqrtf(block_reduce_256(s, red) / (float)DMODEL + EPSN);

  s = 0.f;
  for (int j = threadIdx.x; j < DMODEL; j += 256) {
    const float a = y[j];
    const float dp = 2.f * (a * inv * sc[j] - v[j]);
    s += dp * sc[j] * a;
  }
  const float s2 = block_reduce_256(s, red);

  const float inv3_over_d = inv * inv * inv / (float)DMODEL;
  for (int j = threadIdx.x; j < DMODEL; j += 256) {
    const float a  = y[j];
    const float dp = 2.f * (a * inv * sc[j] - v[j]);
    dYw[(size_t)t * DMODEL + j] = wgt * (inv * sc[j] * dp - inv3_over_d * a * s2);
    dSc[(size_t)t * DMODEL + j] = wgt * (dp * a * inv);
  }
}

// out = rmsnorm(rmsnorm(Y2pre, scu), out_scale)
__global__ void final_norm_kernel(const float* __restrict__ Y2pre,
                                  const float* __restrict__ scu,
                                  const float* __restrict__ osc,
                                  float* __restrict__ out)
{
  __shared__ float red[256];
  const int t = blockIdx.x;
  const float* y = Y2pre + (size_t)t * DMODEL;

  float s = 0.f;
  for (int j = threadIdx.x; j < DMODEL; j += 256) { const float a = y[j]; s += a * a; }
  const float inv1 = rsqrtf(block_reduce_256(s, red) / (float)DMODEL + EPSN);

  s = 0.f;
  for (int j = threadIdx.x; j < DMODEL; j += 256) {
    const float p = y[j] * inv1 * scu[j];
    s += p * p;
  }
  const float inv2 = rsqrtf(block_reduce_256(s, red) / (float)DMODEL + EPSN);

  for (int j = threadIdx.x; j < DMODEL; j += 256) {
    const float p = y[j] * inv1 * scu[j];
    out[(size_t)t * DMODEL + j] = p * inv2 * osc[j];
  }
}

// ---------------------------------------------------------------------------
// Launcher
// ---------------------------------------------------------------------------
extern "C" void kernel_launch(void* const* d_in, const int* in_sizes, int n_in,
                              void* d_out, int out_size, void* d_ws, size_t ws_size,
                              hipStream_t stream)
{
  const float* x    = (const float*)d_in[0];
  const float* ckw  = (const float*)d_in[1];
  const float* ckb  = (const float*)d_in[2];
  const float* cvw  = (const float*)d_in[3];
  const float* cvb  = (const float*)d_in[4];
  const float* rks  = (const float*)d_in[5];
  const float* rvs  = (const float*)d_in[6];
  const float* Wk   = (const float*)d_in[7];
  const float* bk   = (const float*)d_in[8];
  const float* Wv   = (const float*)d_in[9];
  const float* bv   = (const float*)d_in[10];
  const float* w1   = (const float*)d_in[11];
  const float* b1   = (const float*)d_in[12];
  const float* w2   = (const float*)d_in[13];
  const float* b2   = (const float*)d_in[14];
  const float* msc  = (const float*)d_in[15];
  const float* osc  = (const float*)d_in[16];
  float* out        = (float*)d_out;

  float* ws = (float*)d_ws;
  size_t off = 0;
  auto alloc = [&](size_t n) { float* p = ws + off; off += n; return p; };
  const size_t TD  = (size_t)TSEQ * DMODEL;   // 200704
  const size_t TH  = (size_t)TSEQ * HDIM;     // 32768
  const size_t W1N = (size_t)DMODEL * HDIM;   // 1605632

  float* nk    = alloc(TD);
  float* nv    = alloc(TD);
  float* keys  = alloc(TD);
  float* vals  = alloc(TD);
  float* Hpre  = alloc(TH);
  float* H     = alloc(TH);
  float* Ypre  = alloc(TD);
  float* dYw   = alloc(TD);
  float* dSc   = alloc(TD);
  float* dH    = alloc(TH);
  float* dG    = alloc(TH);
  float* aggw1 = alloc(W1N);
  float* aggw2 = alloc(W1N);
  float* aggb1 = alloc(HDIM);
  float* aggb2 = alloc(DMODEL);
  float* aggsc = alloc(DMODEL);
  float* w1u   = alloc(W1N);
  float* w2u   = alloc(W1N);
  float* b1u   = alloc(HDIM);
  float* b2u   = alloc(DMODEL);
  float* scu   = alloc(DMODEL);
  float* H2pre = alloc(TH);
  float* H2    = alloc(TH);
  float* Y2pre = alloc(TD);
  (void)ws_size; (void)n_in; (void)in_sizes; (void)out_size;

  // weights[t] = ETA0 * alpha^(T-1) (constant across tokens), times 1e-4 agg factor
  const float wgt = 1e-4f * 0.1f * powf(0.9f, (float)(TSEQ - 1));

  const dim3 blk128(128);
  auto grd = [](int N, int M) { return dim3((unsigned)((N + 63) / 64), (unsigned)(M / 16)); };

  // Stage 1: conv + rmsnorm  -> nk, nv  (T x D)
  conv_rms_kernel<<<TSEQ, 256, 0, stream>>>(x, ckw, ckb, rks, cvw, cvb, rvs, nk, nv);

  // Stage 2: keys/vals projections, batched over blockIdx.z (concurrent HBM streams)
  gemm_nn_tdm<<<dim3(DMODEL / BN, 1, 2), blk128, 0, stream>>>(
      nk, Wk, bk, keys, nv, Wv, bv, vals, DMODEL, DMODEL, DMODEL, DMODEL, DMODEL);

  // Stage 3: memory-MLP forward
  gemm_nn_tdm<<<dim3(HDIM / BN, 1, 1), blk128, 0, stream>>>(
      keys, w1, b1, Hpre, keys, w1, b1, Hpre, HDIM, DMODEL, DMODEL, HDIM, HDIM);
  gelu_kernel<<<(int)((TH + 255) / 256), 256, 0, stream>>>(Hpre, H, (int)TH);
  gemm_nn_tdm<<<dim3(DMODEL / BN, 1, 1), blk128, 0, stream>>>(
      H, w2, b2, Ypre, H, w2, b2, Ypre, DMODEL, HDIM, HDIM, DMODEL, DMODEL);

  // Stage 4: backward (per-token weight folded into dYw)
  rms_bwd_kernel<<<TSEQ, 256, 0, stream>>>(Ypre, vals, msc, dYw, dSc, wgt);
  colsum_kernel<<<(DMODEL + 255) / 256, 256, 0, stream>>>(dYw, aggb2, TSEQ, DMODEL);
  colsum_kernel<<<(DMODEL + 255) / 256, 256, 0, stream>>>(dSc, aggsc, TSEQ, DMODEL);

  gemm_wmma_bf16<false, true><<<grd(HDIM, TSEQ), blk128, 0, stream>>>(   // dH = dYw @ W2^T
      dYw, w2, nullptr, dH, TSEQ, HDIM, DMODEL, DMODEL, DMODEL, HDIM);
  dgelu_mul_kernel<<<(int)((TH + 255) / 256), 256, 0, stream>>>(dH, Hpre, dG, (int)TH);
  colsum_kernel<<<(HDIM + 255) / 256, 256, 0, stream>>>(dG, aggb1, TSEQ, HDIM);

  gemm_wmma_bf16<true, false><<<grd(DMODEL, HDIM), blk128, 0, stream>>>( // aggw2 = H^T @ dYw
      H, dYw, nullptr, aggw2, HDIM, DMODEL, TSEQ, HDIM, DMODEL, DMODEL);
  gemm_wmma_bf16<true, false><<<grd(HDIM, DMODEL), blk128, 0, stream>>>( // aggw1 = keys^T @ dG
      keys, dG, nullptr, aggw1, DMODEL, HDIM, TSEQ, DMODEL, HDIM, HDIM);

  // Stage 5: parameter update
  sub_kernel<<<(int)((W1N + 255) / 256), 256, 0, stream>>>(w1, aggw1, w1u, (int)W1N);
  sub_kernel<<<(int)((W1N + 255) / 256), 256, 0, stream>>>(w2, aggw2, w2u, (int)W1N);
  sub_kernel<<<(HDIM + 255) / 256, 256, 0, stream>>>(b1, aggb1, b1u, HDIM);
  sub_kernel<<<(DMODEL + 255) / 256, 256, 0, stream>>>(b2, aggb2, b2u, DMODEL);
  sub_kernel<<<(DMODEL + 255) / 256, 256, 0, stream>>>(msc, aggsc, scu, DMODEL);

  // Stage 6: forward with updated memory + double rmsnorm output
  gemm_nn_tdm<<<dim3(HDIM / BN, 1, 1), blk128, 0, stream>>>(
      keys, w1u, b1u, H2pre, keys, w1u, b1u, H2pre, HDIM, DMODEL, DMODEL, HDIM, HDIM);
  gelu_kernel<<<(int)((TH + 255) / 256), 256, 0, stream>>>(H2pre, H2, (int)TH);
  gemm_nn_tdm<<<dim3(DMODEL / BN, 1, 1), blk128, 0, stream>>>(
      H2, w2u, b2u, Y2pre, H2, w2u, b2u, Y2pre, DMODEL, HDIM, HDIM, DMODEL, DMODEL);
  final_norm_kernel<<<TSEQ, 256, 0, stream>>>(Y2pre, scu, osc, out);
}
